// KoopmanTwin_8306466750913
// MI455X (gfx1250) — matrix operand order repaired and checked by reading, not compiled, not run on hardware
//
#include <hip/hip_runtime.h>
#include <hip/hip_bf16.h>

// ---------------------------------------------------------------------------
// KoopmanTwin rollout on gfx1250 (MI455X).  All GEMMs via v_wmma_f32_16x16x32_f16
// (f16 operands, fp32 accumulate).  LayerNorm / GRU gates / residuals in fp32.
// GRU x-tiles are streamed into LDS with the Tensor Data Mover (double-buffered,
// overlapped with WMMA compute, synchronized via TENSORcnt).
// ---------------------------------------------------------------------------

typedef __attribute__((ext_vector_type(16))) _Float16 v16h;
typedef __attribute__((ext_vector_type(8)))  _Float16 v8h;
typedef __attribute__((ext_vector_type(8)))  float    v8f;
typedef __attribute__((ext_vector_type(4)))  unsigned int u32x4;
typedef __attribute__((ext_vector_type(8)))  unsigned int u32x8;

#define Bn   1024
#define Tn   128
#define TF   64
#define Sn   64
#define Cn   16
#define Ln   256
#define Hn   128

static __device__ __forceinline__ float gelu_f(float x) {
    return 0.5f * x * (1.0f + erff(x * 0.70710678118654752f));
}
static __device__ __forceinline__ float sigmoid_f(float x) {
    return 1.0f / (1.0f + __expf(-x));
}

// A-fragment (16x32 f16, row-major source, row stride lda halves).
// lanes 0-15 row m=lane, K 0..15; lanes 16-31 same rows, K 16..31 (contiguous,
// matching load_fragB).  Vector element e -> K=kb+e.
static __device__ __forceinline__ v16h load_fragA(const _Float16* A, int lda, int lane) {
    int m  = lane & 15;
    int kb = (lane >> 4) << 4;              // 0 or 16
    const _Float16* p = A + m * lda + kb;
    v8h lo = *(const v8h*)(p);
    v8h hi = *(const v8h*)(p + 8);
    v16h r;
#pragma unroll
    for (int i = 0; i < 8; ++i) { r[i] = lo[i]; r[i + 8] = hi[i]; }
    return r;
}

// B-fragment (32x16 f16).  W row-major [N,Kp]; we compute X@W^T so B[k][n]=W[n][k].
// lane n=lane&15; lanes 0-15 hold K=0..15, lanes 16-31 hold K=16..31, contiguous.
static __device__ __forceinline__ v16h load_fragB(const _Float16* W, int ldw, int lane) {
    int n  = lane & 15;
    int kb = (lane >> 4) << 4;              // 0 or 16
    return *(const v16h*)(W + n * ldw + kb);
}

// C = act(A[M,Kp] @ W[N,Kp]^T + bias).  A in LDS f16, W in global f16.
// Writes f32 to Cf (if non-null) else f16 to Ch.  M,N mult of 16; Kp mult of 32.
// Must be called by all threads of the block (WMMA needs EXEC all-ones).
static __device__ __forceinline__ void gemm_xwT(
    const _Float16* A, int lda, const _Float16* W, int ldw,
    const float* bias, int M, int N, int Kp, int act,
    float* Cf, _Float16* Ch, int ldc)
{
    const int lane = threadIdx.x & 31;
    const int wave = threadIdx.x >> 5;
    const int nw   = blockDim.x >> 5;
    const int tM = M >> 4, tN = N >> 4;
    for (int tile = wave; tile < tM * tN; tile += nw) {
        int tm = tile % tM, tn = tile / tM;
        v8f acc = {};
        const _Float16* Ab = A + tm * 16 * lda;
        const _Float16* Wb = W + tn * 16 * ldw;
        for (int k = 0; k < Kp; k += 32) {
            v16h af = load_fragA(Ab + k, lda, lane);
            v16h bf = load_fragB(Wb + k, ldw, lane);
            acc = __builtin_amdgcn_wmma_f32_16x16x32_f16(
                false, af, false, bf, (short)0, acc, false, false);
        }
        int n  = tn * 16 + (lane & 15);
        int m0 = tm * 16 + ((lane >> 4) << 3);
        float bv = bias ? bias[n] : 0.0f;
#pragma unroll
        for (int v = 0; v < 8; ++v) {
            float x = acc[v] + bv;
            if (act) x = gelu_f(x);
            if (Cf) Cf[(m0 + v) * ldc + n] = x;
            else    Ch[(m0 + v) * ldc + n] = (_Float16)x;
        }
    }
}

// ---- Tensor Data Mover: 1-D contiguous global->LDS copy (ndw dwords) -------
// D# group0: count=1, lds_addr, 57-bit global addr, type=2 ("image").
// D# group1: data_size=4B, tensor_dim0=tile_dim0=ndw, tensor_dim1=tile_dim1=1.
static __device__ __forceinline__ void tdm_load_1d(const void* gsrc,
                                                   unsigned lds_off, int ndw) {
    unsigned long long ga = (unsigned long long)(uintptr_t)gsrc;
    u32x4 g0;
    g0[0] = 1u;                                           // count=1 (valid)
    g0[1] = lds_off;                                      // LDS byte address
    g0[2] = (unsigned)ga;                                 // global addr [31:0]
    g0[3] = ((unsigned)(ga >> 32) & 0x01FFFFFFu) | 0x80000000u; // [56:32] | type=2
    u32x8 g1;
    g1[0] = (2u << 16);                                   // data_size=4B
    g1[1] = ((unsigned)ndw & 0xFFFFu) << 16;              // tensor_dim0 [15:0]
    g1[2] = (((unsigned)ndw >> 16) & 0xFFFFu) | (1u << 16); // dim0 hi | tensor_dim1=1
    g1[3] = ((unsigned)ndw & 0xFFFFu) << 16;              // tile_dim0
    g1[4] = 1u;                                           // tile_dim1=1, tile_dim2=0
    g1[5] = (unsigned)ndw;                                // tensor_dim0_stride lo
    g1[6] = 0u;
    g1[7] = 0u;
    asm volatile("tensor_load_to_lds %0, %1" :: "s"(g0), "s"(g1) : "memory");
}

// ------------------------------ prep kernels -------------------------------

__global__ void k_zero_h(_Float16* p, int n) {
    for (int i = blockIdx.x * blockDim.x + threadIdx.x; i < n; i += gridDim.x * blockDim.x)
        p[i] = (_Float16)0.0f;
}

__global__ void k_copy_w(const float* __restrict__ src, _Float16* __restrict__ dst,
                         int rows, int cols, int dstLd, int dstOff) {
    int n = rows * cols;
    for (int i = blockIdx.x * blockDim.x + threadIdx.x; i < n; i += gridDim.x * blockDim.x) {
        int r = i / cols, c = i % cols;
        dst[r * dstLd + dstOff + c] = (_Float16)src[i];
    }
}

// --------------------------- encoder projection ----------------------------
// rows m = t*B + b (time-major).  out f16 xbuf0[m*256 + n]
__global__ void k_encproj(const float* __restrict__ xh, const float* __restrict__ uh,
                          const _Float16* __restrict__ W, const float* __restrict__ bias,
                          _Float16* __restrict__ xbuf) {
    __shared__ _Float16 AT[16][96];
    int m0 = blockIdx.x * 16, tid = threadIdx.x;
    for (int i = tid; i < 16 * 96; i += blockDim.x) {
        int r = i / 96, c = i % 96;
        int m = m0 + r, b = m & (Bn - 1), t = m >> 10;
        float v = 0.0f;
        if (c < Sn)           v = xh[((size_t)b * Tn + t) * Sn + c];
        else if (c < Sn + Cn) v = uh[((size_t)b * Tn + t) * Cn + (c - Sn)];
        AT[r][c] = (_Float16)v;
    }
    __syncthreads();
    gemm_xwT(&AT[0][0], 96, W, 96, bias, 16, Ln, 96, 0,
             nullptr, xbuf + (size_t)m0 * Ln, Ln);
}

// ------------------------------- GRU layer ---------------------------------
// grid (B/16, 2): y = direction.  xin/xout f16 [T, B, 256] time-major.
// x_t tiles (16 rows x 512B, contiguous 8KB) streamed via TDM, double-buffered.
__global__ void k_gru(const _Float16* __restrict__ xin, _Float16* __restrict__ xout,
                      const _Float16* wihF, const _Float16* whhF,
                      const float* bihF, const float* bhhF,
                      const _Float16* wihB, const _Float16* whhB,
                      const float* bihB, const float* bhhB) {
    __shared__ _Float16 Xb[2][16][256];  // TDM-filled x tiles (double buffer)
    __shared__ _Float16 Hb[16][128];     // hidden state (f16)
    __shared__ float    GX[16][384];     // x-path gate pre-activations (f32)
    __shared__ _Float16 GH[16][384];     // h-path gate pre-activations (f16)
    int dir = blockIdx.y, rb = blockIdx.x * 16, tid = threadIdx.x;
    const _Float16* wih = dir ? wihB : wihF;
    const _Float16* whh = dir ? whhB : whhF;
    const float*    bih = dir ? bihB : bihF;
    const float*    bhh = dir ? bhhB : bhhF;

    for (int i = tid; i < 16 * Hn; i += blockDim.x)
        Hb[i >> 7][i & 127] = (_Float16)0.0f;

    const int ndw = (16 * Ln * 2) / 4;   // 2048 dwords = 8KB tile
    // prologue: fetch first timestep into buffer 0 (wave 0 issues; TDM ignores EXEC)
    if (tid < 32) {
        int t0 = dir ? (Tn - 1) : 0;
        tdm_load_1d(xin + ((size_t)t0 * Bn + rb) * Ln,
                    (unsigned)(uintptr_t)&Xb[0][0][0], ndw);
    }

    for (int tt = 0; tt < Tn; ++tt) {
        int t   = dir ? (Tn - 1 - tt) : tt;
        int buf = tt & 1;
        if (tid < 32) __builtin_amdgcn_s_wait_tensorcnt(0);   // Xb[buf] ready
        __syncthreads();
        if (tt + 1 < Tn) {                                    // prefetch next tile
            int tnext = dir ? (Tn - 2 - tt) : (tt + 1);
            if (tid < 32)
                tdm_load_1d(xin + ((size_t)tnext * Bn + rb) * Ln,
                            (unsigned)(uintptr_t)&Xb[buf ^ 1][0][0], ndw);
        }
        gemm_xwT(&Xb[buf][0][0], 256, wih, 256, bih, 16, 3 * Hn, 256, 0,
                 &GX[0][0], nullptr, 384);
        gemm_xwT(&Hb[0][0], 128, whh, 128, bhh, 16, 3 * Hn, 128, 0,
                 nullptr, &GH[0][0], 384);
        __syncthreads();
        for (int i = tid; i < 16 * Hn; i += blockDim.x) {
            int r = i >> 7, j = i & 127;
            float rg = sigmoid_f(GX[r][j]       + (float)GH[r][j]);
            float zg = sigmoid_f(GX[r][128 + j] + (float)GH[r][128 + j]);
            float nn = tanhf(GX[r][256 + j] + rg * (float)GH[r][256 + j]);
            float hp = (float)Hb[r][j];
            float hn = (1.0f - zg) * nn + zg * hp;
            Hb[r][j] = (_Float16)hn;
            xout[((size_t)t * Bn + rb + r) * Ln + dir * Hn + j] = (_Float16)hn;
        }
        __syncthreads();
    }
}

// ------------------------------ encoder head -------------------------------
__global__ void k_head(const _Float16* __restrict__ xbuf, const float* __restrict__ xh,
                       const float* lnw, const float* lnb,
                       const _Float16* Wfc1, const float* bfc1,
                       const _Float16* Wfc2, const float* bfc2,
                       float* __restrict__ zbuf, float* __restrict__ xprev) {
    __shared__ float    Hf[16][256];
    __shared__ _Float16 HT[16][256];
    __shared__ _Float16 HT2[16][256];
    int rb = blockIdx.x * 16, tid = threadIdx.x;
    for (int i = tid; i < 16 * Ln; i += blockDim.x) {
        int r = i >> 8, c = i & 255;
        Hf[r][c] = (float)xbuf[((size_t)(Tn - 1) * Bn + rb + r) * Ln + c];
    }
    __syncthreads();
    if (tid < 16) {
        int r = tid; float m = 0.0f, v = 0.0f;
        for (int c = 0; c < Ln; ++c) m += Hf[r][c];
        m *= (1.0f / Ln);
        for (int c = 0; c < Ln; ++c) { float d = Hf[r][c] - m; v += d * d; }
        v *= (1.0f / Ln);
        float inv = rsqrtf(v + 1e-5f);
        for (int c = 0; c < Ln; ++c)
            HT[r][c] = (_Float16)((Hf[r][c] - m) * inv * lnw[c] + lnb[c]);
    }
    __syncthreads();
    gemm_xwT(&HT[0][0],  256, Wfc1, 256, bfc1, 16, Ln, 256, 1, nullptr, &HT2[0][0], 256);
    __syncthreads();
    gemm_xwT(&HT2[0][0], 256, Wfc2, 256, bfc2, 16, Ln, 256, 0, &Hf[0][0], nullptr, 256);
    __syncthreads();
    for (int i = tid; i < 16 * Ln; i += blockDim.x) {
        int r = i >> 8, c = i & 255;
        zbuf[(size_t)(rb + r) * Ln + c] = Hf[r][c];
    }
    for (int i = tid; i < 16 * Sn; i += blockDim.x) {
        int r = i >> 6, s = i & 63;
        xprev[(size_t)(rb + r) * Sn + s] = xh[((size_t)(rb + r) * Tn + (Tn - 1)) * Sn + s];
    }
}

// ------------------------------ rollout step -------------------------------
struct RolloutArgs {
    const float* u_future;
    float* zbuf; float* xprev; float* out;
    const _Float16 *W_AB, *W_d1, *W_d2, *W_d3, *W_e1, *W_e2, *W_e3;
    const float *ln_lin_w, *ln_lin_b, *dyn_b1, *dyn_b2, *dyn_b3;
    const float *ln_res_w, *ln_res_b, *dec_ln_w, *dec_ln_b;
    const float *dec_b1, *dec_b2, *dec_b3;
    int t;
};

__global__ void k_rollout(RolloutArgs a) {
    __shared__ _Float16 IOT[16][320];                  // (z,u) K=288 then (zn,xp) K=320
    __shared__ __align__(16) unsigned char SCR[16 * 512 * 2]; // R1 f16[16][512] / R3 f32[16][256]
    __shared__ _Float16 R2[16][256];
    __shared__ float    LIN[16][256];
    __shared__ float    XP[16][64];
    _Float16* R1 = (_Float16*)SCR;
    float*    R3 = (float*)SCR;
    int rb = blockIdx.x * 16, tid = threadIdx.x, t = a.t;

    for (int i = tid; i < 16 * 320; i += blockDim.x) {
        int r = i / 320, c = i % 320, row = rb + r;
        float v = 0.0f;
        if (c < Ln)           v = a.zbuf[(size_t)row * Ln + c];
        else if (c < Ln + Cn) v = a.u_future[((size_t)row * TF + t) * Cn + (c - Ln)];
        IOT[r][c] = (_Float16)v;
    }
    for (int i = tid; i < 16 * Sn; i += blockDim.x)
        XP[i >> 6][i & 63] = a.xprev[(size_t)(rb + (i >> 6)) * Sn + (i & 63)];
    __syncthreads();

    // lin = z@A^T + u@B^T  (fused weight, K=288) ; r1 = gelu(concat@dyn1^T)
    gemm_xwT(&IOT[0][0], 320, a.W_AB, 288, nullptr, 16, Ln, 288, 0, &LIN[0][0], nullptr, 256);
    gemm_xwT(&IOT[0][0], 320, a.W_d1, 288, a.dyn_b1, 16, 512, 288, 1, nullptr, R1, 512);
    __syncthreads();

    if (tid < 16) {                                    // LN(lin) in place
        int r = tid; float m = 0.0f, v = 0.0f;
        for (int c = 0; c < Ln; ++c) m += LIN[r][c];
        m *= (1.0f / Ln);
        for (int c = 0; c < Ln; ++c) { float d = LIN[r][c] - m; v += d * d; }
        v *= (1.0f / Ln);
        float inv = rsqrtf(v + 1e-5f);
        for (int c = 0; c < Ln; ++c)
            LIN[r][c] = (LIN[r][c] - m) * inv * a.ln_lin_w[c] + a.ln_lin_b[c];
    }
    gemm_xwT(R1, 512, a.W_d2, 512, a.dyn_b2, 16, Ln, 512, 1, nullptr, &R2[0][0], 256);
    __syncthreads();
    gemm_xwT(&R2[0][0], 256, a.W_d3, 256, a.dyn_b3, 16, Ln, 256, 0, R3, nullptr, 256);
    __syncthreads();

    if (tid < 16) {                                    // z_new + dec_ln -> IOT
        int r = tid, row = rb + r;
        float m = 0.0f, v = 0.0f;
        for (int c = 0; c < Ln; ++c) m += R3[r * 256 + c];
        m *= (1.0f / Ln);
        for (int c = 0; c < Ln; ++c) { float d = R3[r * 256 + c] - m; v += d * d; }
        v *= (1.0f / Ln);
        float inv = rsqrtf(v + 1e-5f);
        for (int c = 0; c < Ln; ++c) {
            float nr = (R3[r * 256 + c] - m) * inv * a.ln_res_w[c] + a.ln_res_b[c];
            float z  = LIN[r][c] + 0.3f * nr;
            a.zbuf[(size_t)row * Ln + c] = z;
            LIN[r][c] = z;
        }
        m = 0.0f; v = 0.0f;
        for (int c = 0; c < Ln; ++c) m += LIN[r][c];
        m *= (1.0f / Ln);
        for (int c = 0; c < Ln; ++c) { float d = LIN[r][c] - m; v += d * d; }
        v *= (1.0f / Ln);
        float inv2 = rsqrtf(v + 1e-5f);
        for (int c = 0; c < Ln; ++c)
            IOT[r][c] = (_Float16)((LIN[r][c] - m) * inv2 * a.dec_ln_w[c] + a.dec_ln_b[c]);
        for (int s = 0; s < Sn; ++s) IOT[r][256 + s] = (_Float16)XP[r][s];
    }
    __syncthreads();

    gemm_xwT(&IOT[0][0], 320, a.W_e1, 320, a.dec_b1, 16, 512, 320, 1, nullptr, R1, 512);
    __syncthreads();
    gemm_xwT(R1, 512, a.W_e2, 512, a.dec_b2, 16, Ln, 512, 1, nullptr, &R2[0][0], 256);
    __syncthreads();
    gemm_xwT(&R2[0][0], 256, a.W_e3, 256, a.dec_b3, 16, Sn, 256, 0, R3, nullptr, 64);
    __syncthreads();

    for (int i = tid; i < 16 * Sn; i += blockDim.x) {
        int r = i >> 6, s = i & 63, row = rb + r;
        float xp = XP[r][s] + R3[r * 64 + s];
        a.xprev[(size_t)row * Sn + s] = xp;
        a.out[((size_t)row * TF + t) * Sn + s] = xp;    // [B, T_FUT, S]
    }
}

// ------------------------------- host side ---------------------------------

extern "C" void kernel_launch(void* const* d_in, const int* in_sizes, int n_in,
                              void* d_out, int out_size, void* d_ws, size_t ws_size,
                              hipStream_t stream) {
    (void)in_sizes; (void)n_in; (void)out_size; (void)ws_size;
    const float* x_hist   = (const float*)d_in[0];
    const float* u_hist   = (const float*)d_in[1];
    const float* u_future = (const float*)d_in[2];
    // param indices (dict insertion order)
    const float* enc_proj_w = (const float*)d_in[3];
    const float* enc_proj_b = (const float*)d_in[4];
    const float* gw[4][4];  // [l0f,l0b,l1f,l1b][wih,whh,bih,bhh]
    for (int g = 0; g < 4; ++g)
        for (int k = 0; k < 4; ++k) gw[g][k] = (const float*)d_in[5 + g * 4 + k];
    const float* enc_ln_w = (const float*)d_in[21];
    const float* enc_ln_b = (const float*)d_in[22];
    const float* fc1_w = (const float*)d_in[23];
    const float* fc1_b = (const float*)d_in[24];
    const float* fc2_w = (const float*)d_in[25];
    const float* fc2_b = (const float*)d_in[26];
    const float* A_w   = (const float*)d_in[27];
    const float* B_w   = (const float*)d_in[28];
    const float* ln_lin_w = (const float*)d_in[29];
    const float* ln_lin_b = (const float*)d_in[30];
    const float* dyn_w1 = (const float*)d_in[31];
    const float* dyn_b1 = (const float*)d_in[32];
    const float* dyn_w2 = (const float*)d_in[33];
    const float* dyn_b2 = (const float*)d_in[34];
    const float* dyn_w3 = (const float*)d_in[35];
    const float* dyn_b3 = (const float*)d_in[36];
    const float* ln_res_w = (const float*)d_in[37];
    const float* ln_res_b = (const float*)d_in[38];
    const float* dec_ln_w = (const float*)d_in[39];
    const float* dec_ln_b = (const float*)d_in[40];
    const float* dec_w1 = (const float*)d_in[41];
    const float* dec_b1 = (const float*)d_in[42];
    const float* dec_w2 = (const float*)d_in[43];
    const float* dec_b2 = (const float*)d_in[44];
    const float* dec_w3 = (const float*)d_in[45];
    const float* dec_b3 = (const float*)d_in[46];

    char* ws = (char*)d_ws;
    size_t off = 0;
    auto alloc = [&](size_t bytes) -> void* {
        void* p = ws + off;
        off = (off + bytes + 255) & ~(size_t)255;
        return p;
    };
    _Float16* W_encproj = (_Float16*)alloc((size_t)256 * 96 * 2);
    _Float16* W_wih[4]; _Float16* W_whh[4];
    for (int g = 0; g < 4; ++g) {
        W_wih[g] = (_Float16*)alloc((size_t)384 * 256 * 2);
        W_whh[g] = (_Float16*)alloc((size_t)384 * 128 * 2);
    }
    _Float16* W_fc1 = (_Float16*)alloc((size_t)256 * 256 * 2);
    _Float16* W_fc2 = (_Float16*)alloc((size_t)256 * 256 * 2);
    _Float16* W_AB  = (_Float16*)alloc((size_t)256 * 288 * 2);
    _Float16* W_d1  = (_Float16*)alloc((size_t)512 * 288 * 2);
    _Float16* W_d2  = (_Float16*)alloc((size_t)256 * 512 * 2);
    _Float16* W_d3  = (_Float16*)alloc((size_t)256 * 256 * 2);
    _Float16* W_e1  = (_Float16*)alloc((size_t)512 * 320 * 2);
    _Float16* W_e2  = (_Float16*)alloc((size_t)256 * 512 * 2);
    _Float16* W_e3  = (_Float16*)alloc((size_t)64  * 256 * 2);
    size_t wbytes = off;                                   // end of f16 weight arena
    _Float16* xbuf0 = (_Float16*)alloc((size_t)Tn * Bn * Ln * 2);
    _Float16* xbuf1 = (_Float16*)alloc((size_t)Tn * Bn * Ln * 2);
    float* zbuf  = (float*)alloc((size_t)Bn * Ln * 4);
    float* xprev = (float*)alloc((size_t)Bn * Sn * 4);

    // ---- weight prep (zero + padded f16 copies) ----
    k_zero_h<<<512, 256, 0, stream>>>(W_encproj, (int)(wbytes / 2));
    k_copy_w<<<128, 256, 0, stream>>>(enc_proj_w, W_encproj, 256, 80, 96, 0);
    for (int g = 0; g < 4; ++g) {
        k_copy_w<<<256, 256, 0, stream>>>(gw[g][0], W_wih[g], 384, 256, 256, 0);
        k_copy_w<<<128, 256, 0, stream>>>(gw[g][1], W_whh[g], 384, 128, 128, 0);
    }
    k_copy_w<<<128, 256, 0, stream>>>(fc1_w, W_fc1, 256, 256, 256, 0);
    k_copy_w<<<128, 256, 0, stream>>>(fc2_w, W_fc2, 256, 256, 256, 0);
    k_copy_w<<<128, 256, 0, stream>>>(A_w,   W_AB,  256, 256, 288, 0);
    k_copy_w<<<16,  256, 0, stream>>>(B_w,   W_AB,  256, 16,  288, 256);
    k_copy_w<<<256, 256, 0, stream>>>(dyn_w1, W_d1, 512, 272, 288, 0);
    k_copy_w<<<256, 256, 0, stream>>>(dyn_w2, W_d2, 256, 512, 512, 0);
    k_copy_w<<<128, 256, 0, stream>>>(dyn_w3, W_d3, 256, 256, 256, 0);
    k_copy_w<<<256, 256, 0, stream>>>(dec_w1, W_e1, 512, 320, 320, 0);
    k_copy_w<<<256, 256, 0, stream>>>(dec_w2, W_e2, 256, 512, 512, 0);
    k_copy_w<<<32,  256, 0, stream>>>(dec_w3, W_e3, 64,  256, 256, 0);

    // ---- encoder ----
    k_encproj<<<(Bn * Tn) / 16, 128, 0, stream>>>(x_hist, u_hist, W_encproj, enc_proj_b, xbuf0);
    k_gru<<<dim3(Bn / 16, 2), 128, 0, stream>>>(xbuf0, xbuf1,
        W_wih[0], W_whh[0], gw[0][2], gw[0][3],
        W_wih[1], W_whh[1], gw[1][2], gw[1][3]);
    k_gru<<<dim3(Bn / 16, 2), 128, 0, stream>>>(xbuf1, xbuf0,
        W_wih[2], W_whh[2], gw[2][2], gw[2][3],
        W_wih[3], W_whh[3], gw[3][2], gw[3][3]);
    k_head<<<Bn / 16, 128, 0, stream>>>(xbuf0, x_hist, enc_ln_w, enc_ln_b,
        W_fc1, fc1_b, W_fc2, fc2_b, zbuf, xprev);

    // ---- rollout ----
    RolloutArgs ra;
    ra.u_future = u_future; ra.zbuf = zbuf; ra.xprev = xprev; ra.out = (float*)d_out;
    ra.W_AB = W_AB; ra.W_d1 = W_d1; ra.W_d2 = W_d2; ra.W_d3 = W_d3;
    ra.W_e1 = W_e1; ra.W_e2 = W_e2; ra.W_e3 = W_e3;
    ra.ln_lin_w = ln_lin_w; ra.ln_lin_b = ln_lin_b;
    ra.dyn_b1 = dyn_b1; ra.dyn_b2 = dyn_b2; ra.dyn_b3 = dyn_b3;
    ra.ln_res_w = ln_res_w; ra.ln_res_b = ln_res_b;
    ra.dec_ln_w = dec_ln_w; ra.dec_ln_b = dec_ln_b;
    ra.dec_b1 = dec_b1; ra.dec_b2 = dec_b2; ra.dec_b3 = dec_b3;
    for (int t = 0; t < TF; ++t) {
        ra.t = t;
        k_rollout<<<Bn / 16, 128, 0, stream>>>(ra);
    }
}
